// TransformerBlock_36550171689207
// MI455X (gfx1250) — compile-verified, matrix operand order checked by reading, not run on hardware
//
#include <hip/hip_runtime.h>

// ---------------------------------------------------------------------------
// Transformer block for MI455X (gfx1250): bf16 WMMA everywhere,
// TDM (tensor_load_to_lds) staging for GEMM tiles, transposed-V flash attn.
// B=4, S=2048, D=1024, H=16, DH=64
// ---------------------------------------------------------------------------

#define B_ 4
#define S_ 2048
#define D_ 1024
#define H_ 16
#define DH_ 64
#define BS_ (B_ * S_)
#define EPS_ 1e-5f

typedef unsigned short u16;
typedef __attribute__((ext_vector_type(16))) __bf16 bf16x16;
typedef __attribute__((ext_vector_type(8)))  float  f32x8;
typedef __attribute__((ext_vector_type(4)))  unsigned int u32x4;
typedef __attribute__((ext_vector_type(4)))  unsigned int uint32x4;
typedef __attribute__((ext_vector_type(8)))  int int32x8;
typedef __attribute__((ext_vector_type(4)))  int int32x4;

#if defined(__has_builtin)
#if __has_builtin(__builtin_amdgcn_tensor_load_to_lds) && \
    __has_builtin(__builtin_amdgcn_s_wait_tensorcnt)
#define USE_TDM 1
#endif
#endif
#ifndef USE_TDM
#define USE_TDM 0
#endif

union Frag {
    bf16x16 v;
    u16     u[16];
    u32x4   q[2];
};

union Pack8 {
    u16   a[8];
    u32x4 q;
};

__device__ __forceinline__ f32x8 wmma_bf16(bf16x16 a, bf16x16 b, f32x8 c) {
    return __builtin_amdgcn_wmma_f32_16x16x32_bf16(false, a, false, b,
                                                   (short)0, c, false, false);
}

__device__ __forceinline__ u16 f32_to_bf16(float f) {
    unsigned int u = __float_as_uint(f);
    unsigned int r = (u + 0x7FFFu + ((u >> 16) & 1u)) >> 16;
    return (u16)r;
}

__device__ __forceinline__ float gelu_tanh_f(float v) {
    const float c = 0.7978845608028654f;  // sqrt(2/pi)
    float u = c * (v + 0.044715f * v * v * v);
    return 0.5f * v * (1.0f + tanhf(u));
}

#if USE_TDM
// Issue a TDM 2D tile load: 32(K-elems) x 128(rows) bf16 tile, row stride
// `strideElems`, destination LDS rows padded to 40 elements (pad 4 DW every
// 16 DW). Must be executed by one wave; completion via TENSORcnt.
__device__ __forceinline__ void tdm_load_tile(const u16* gsrc,
                                              unsigned lds_addr,
                                              int strideElems) {
    const unsigned long long ga = (unsigned long long)(size_t)gsrc;
    uint32x4 g0 = {
        1u,                                   // count=1, user descriptor
        lds_addr,                             // LDS byte address
        (unsigned)(ga & 0xffffffffu),         // global addr [31:0]
        (unsigned)(ga >> 32) | 0x80000000u    // global addr [56:32] | type=2
    };
    const unsigned dim0 = 1u << 20, dim1 = 1u << 20;  // huge: no OOB clipping
    int32x8 g1 = {
        (int)((1u << 16) |        // data_size = 2 bytes
              (1u << 20) |        // pad_enable
              (3u << 22) |        // pad_interval: 16 DWORDs (one 64B row)
              (3u << 25)),        // pad_amount: 4 DWORDs (16B) -> 80B rows
        (int)((dim0 & 0xffffu) << 16),
        (int)(((dim0 >> 16) & 0xffffu) | ((dim1 & 0xffffu) << 16)),
        (int)(((dim1 >> 16) & 0xffffu) | (32u << 16)),  // tile_dim0 = 32
        (int)128,                                       // tile_dim1 = 128
        (int)strideElems,                               // tensor_dim0_stride
        0, 0
    };
    int32x4 z4 = {0, 0, 0, 0};
    int32x8 z8 = {0, 0, 0, 0, 0, 0, 0, 0};
    __builtin_amdgcn_tensor_load_to_lds(g0, g1, z4, z4, z8, 0);
}
#endif

// ---------------------------------------------------------------------------
// fp32 -> bf16 elementwise convert
// ---------------------------------------------------------------------------
__global__ void cvt_f32_bf16_kernel(const float* __restrict__ in,
                                    u16* __restrict__ out, int n) {
    int i = blockIdx.x * blockDim.x + threadIdx.x;
    int stride = gridDim.x * blockDim.x;
    for (; i < n; i += stride) out[i] = f32_to_bf16(in[i]);
}

// ---------------------------------------------------------------------------
// LayerNorm over D=1024, one block (256 threads = 8 waves) per row.
// ---------------------------------------------------------------------------
__global__ __launch_bounds__(256) void layernorm_bf16_kernel(
    const float* __restrict__ in, const float* __restrict__ scale,
    const float* __restrict__ shift, u16* __restrict__ out) {
    const int row = blockIdx.x;
    const float* p = in + (size_t)row * D_;
    const int tid = threadIdx.x;

    float v[4];
    float s = 0.f, sq = 0.f;
#pragma unroll
    for (int j = 0; j < 4; ++j) {
        v[j] = p[tid + j * 256];
        s += v[j];
        sq += v[j] * v[j];
    }
#pragma unroll
    for (int m = 16; m >= 1; m >>= 1) {
        s += __shfl_xor(s, m, 32);
        sq += __shfl_xor(sq, m, 32);
    }
    __shared__ float rs[8], rq[8], stat[2];
    if ((tid & 31) == 0) { rs[tid >> 5] = s; rq[tid >> 5] = sq; }
    __syncthreads();
    if (tid < 32) {
        float a = (tid < 8) ? rs[tid] : 0.f;
        float b = (tid < 8) ? rq[tid] : 0.f;
#pragma unroll
        for (int m = 4; m >= 1; m >>= 1) {
            a += __shfl_xor(a, m, 32);
            b += __shfl_xor(b, m, 32);
        }
        if (tid == 0) {
            float mean = a * (1.0f / D_);
            float var = b * (1.0f / D_) - mean * mean;
            stat[0] = mean;
            stat[1] = rsqrtf(var + EPS_);
        }
    }
    __syncthreads();
    const float mean = stat[0], rstd = stat[1];
    u16* q = out + (size_t)row * D_;
#pragma unroll
    for (int j = 0; j < 4; ++j) {
        int c = tid + j * 256;
        float y = scale[c] * (v[j] - mean) * rstd + shift[c];
        q[c] = f32_to_bf16(y);
    }
}

// ---------------------------------------------------------------------------
// WMMA GEMM:  C[M,N] = A[M,K] (bf16, row-major) * W[N,K]^T (bf16, row-major)
// 256 threads (8 waves), 128x128 block tile, K-step 32, TDM double-buffered
// LDS staging (fallback: manual staging). Wave tile 32x64 = 2x4 accumulators.
// Epilogue: +bias -> gelu -> +residual -> store f32 / bf16 / bf16-transposed
// (transposed mode writes V^T as [B,H,DH,S] with packed 16B stores).
// ---------------------------------------------------------------------------
template <bool HAS_BIAS, bool DO_GELU, bool HAS_RES, bool OUT_F32,
          bool OUT_BF16, bool OUT_BF16T>
__global__ __launch_bounds__(256) void gemm_bf16_kernel(
    const u16* __restrict__ A, const u16* __restrict__ W,
    const float* __restrict__ bias, const float* __restrict__ res,
    float* __restrict__ outF, u16* __restrict__ outB,
    int M, int N, int K) {
#if USE_TDM
    __shared__ u16 As[2][128][40];
    __shared__ u16 Ws[2][128][40];
#else
    __shared__ u16 As[1][128][40];
    __shared__ u16 Ws[1][128][40];
#endif

    const int tid = threadIdx.x;
    const int lane = tid & 31;
    const int wid = tid >> 5;
    const int half = lane >> 4;
    const int r = lane & 15;
    const int wm = wid & 3;   // M sub-tile (32 rows each)
    const int wn = wid >> 2;  // N sub-tile (64 cols each)
    const int rowBase = blockIdx.y * 128;
    const int colBase = blockIdx.x * 128;

    f32x8 acc[2][4];
#pragma unroll
    for (int i = 0; i < 2; ++i)
#pragma unroll
        for (int j = 0; j < 4; ++j) acc[i][j] = (f32x8){0, 0, 0, 0, 0, 0, 0, 0};

    const int nk = K / 32;

#if USE_TDM
    const u16* Abase = A + (size_t)rowBase * K;
    const u16* Wbase = W + (size_t)colBase * K;
    if (tid < 32) {
        tdm_load_tile(Abase, (unsigned)(size_t)&As[0][0][0], K);
        tdm_load_tile(Wbase, (unsigned)(size_t)&Ws[0][0][0], K);
    }
    for (int kt = 0; kt < nk; ++kt) {
        const int buf = kt & 1;
        if (tid < 32) {
            if (kt + 1 < nk) {
                tdm_load_tile(Abase + (kt + 1) * 32,
                              (unsigned)(size_t)&As[buf ^ 1][0][0], K);
                tdm_load_tile(Wbase + (kt + 1) * 32,
                              (unsigned)(size_t)&Ws[buf ^ 1][0][0], K);
                __builtin_amdgcn_s_wait_tensorcnt(2);  // tile kt complete
            } else {
                __builtin_amdgcn_s_wait_tensorcnt(0);
            }
        }
        __syncthreads();
#else
    const int ldr = tid >> 2;       // 0..63
    const int ldk = (tid & 3) * 8;  // 0,8,16,24 (elements)
    for (int kt = 0; kt < nk; ++kt) {
        const int buf = 0;
        const int k0 = kt * 32;
        u32x4 a0 = *(const u32x4*)(A + (size_t)(rowBase + ldr) * K + k0 + ldk);
        u32x4 a1 = *(const u32x4*)(A + (size_t)(rowBase + ldr + 64) * K + k0 + ldk);
        u32x4 b0 = *(const u32x4*)(W + (size_t)(colBase + ldr) * K + k0 + ldk);
        u32x4 b1 = *(const u32x4*)(W + (size_t)(colBase + ldr + 64) * K + k0 + ldk);
        if (k0 + 32 < K) {
            __builtin_prefetch(A + (size_t)(rowBase + ldr) * K + k0 + 32 + ldk, 0, 0);
            __builtin_prefetch(W + (size_t)(colBase + ldr) * K + k0 + 32 + ldk, 0, 0);
        }
        __syncthreads();
        *(u32x4*)&As[0][ldr][ldk] = a0;
        *(u32x4*)&As[0][ldr + 64][ldk] = a1;
        *(u32x4*)&Ws[0][ldr][ldk] = b0;
        *(u32x4*)&Ws[0][ldr + 64][ldk] = b1;
        __syncthreads();
#endif

        Frag af[2];
#pragma unroll
        for (int i = 0; i < 2; ++i) {
            int rr = wm * 32 + i * 16 + r;
            af[i].q[0] = *(const u32x4*)&As[buf][rr][half * 8];
            af[i].q[1] = *(const u32x4*)&As[buf][rr][16 + half * 8];
        }
        Frag bf_[4];
#pragma unroll
        for (int j = 0; j < 4; ++j) {
            int rr = wn * 64 + j * 16 + r;
            bf_[j].q[0] = *(const u32x4*)&Ws[buf][rr][half * 8];
            bf_[j].q[1] = *(const u32x4*)&Ws[buf][rr][16 + half * 8];
        }
#pragma unroll
        for (int i = 0; i < 2; ++i)
#pragma unroll
            for (int j = 0; j < 4; ++j)
                acc[i][j] = wmma_bf16(af[i].v, bf_[j].v, acc[i][j]);
        __syncthreads();
    }

    // Epilogue. C layout: element e -> row = tileRow + e + 8*half,
    // col = tileCol + (lane & 15).
#pragma unroll
    for (int j = 0; j < 4; ++j) {
        const int colg = colBase + wn * 64 + j * 16 + r;
        float bj = 0.f;
        if (HAS_BIAS) bj = bias[colg];
#pragma unroll
        for (int i = 0; i < 2; ++i) {
            const int rowg0 = rowBase + wm * 32 + i * 16 + half * 8;
            if (OUT_BF16T) {
                // V^T store: [B,H,DH,S]; 8 consecutive rows -> one 16B store.
                Pack8 pk;
#pragma unroll
                for (int e = 0; e < 8; ++e) {
                    float v = acc[i][j][e];
                    if (HAS_BIAS) v += bj;
                    pk.a[e] = f32_to_bf16(v);
                }
                const int bq = rowg0 >> 11;          // / S_
                const int sv = rowg0 & (S_ - 1);
                const size_t idx =
                    (((size_t)bq * H_ + (colg >> 6)) * DH_ + (colg & 63)) * S_ +
                    sv;
                *(u32x4*)(outB + idx) = pk.q;
            } else {
#pragma unroll
                for (int e = 0; e < 8; ++e) {
                    const int rowg = rowg0 + e;
                    float v = acc[i][j][e];
                    if (HAS_BIAS) v += bj;
                    if (DO_GELU) v = gelu_tanh_f(v);
                    if (HAS_RES) v += res[(size_t)rowg * N + colg];
                    if (OUT_F32) outF[(size_t)rowg * N + colg] = v;
                    if (OUT_BF16) outB[(size_t)rowg * N + colg] = f32_to_bf16(v);
                }
            }
        }
    }
}

// ---------------------------------------------------------------------------
// Causal flash attention, bf16 WMMA, f32 softmax state.
// Block = 128 threads (4 waves); each wave owns one 16-query tile.
// 32 keys/step: QK^T (4 wmma), online softmax, P via per-wave LDS re-layout,
// V read directly from the transposed [B,H,DH,S] layout (contiguous 16B
// B-fragment loads), P*V (4 wmma).
// ---------------------------------------------------------------------------
__global__ __launch_bounds__(128) void flash_attn_kernel(
    const u16* __restrict__ Q, const u16* __restrict__ Kb,
    const u16* __restrict__ Vt, u16* __restrict__ Ctx) {
    __shared__ u16 Pl[4][16][40];  // per-wave P tile (16 q x 32 k), padded

    const int tid = threadIdx.x;
    const int w = tid >> 5;
    const int lane = tid & 31;
    const int half = lane >> 4;
    const int r = lane & 15;
    const int b = blockIdx.z;
    const int h = blockIdx.y;
    const int q0 = (blockIdx.x * 4 + w) * 16;

    const size_t headOff = (size_t)b * S_ * D_ + (size_t)h * DH_;
    const u16* Qp = Q + headOff;
    const u16* Kp = Kb + headOff;
    const u16* Vth = Vt + ((size_t)b * H_ + h) * DH_ * S_;  // [DH][S]

    // Q fragments (A layout): row = q0 + r, elements along dh.
    Frag qf[2];
    {
        const u16* Qrow = Qp + (size_t)(q0 + r) * D_;
#pragma unroll
        for (int i = 0; i < 2; ++i) {
            qf[i].q[0] = *(const u32x4*)(Qrow + i * 32 + half * 8);
            qf[i].q[1] = *(const u32x4*)(Qrow + i * 32 + 16 + half * 8);
        }
    }

    float m_r[8], l_r[8];
    f32x8 o[4];
#pragma unroll
    for (int e = 0; e < 8; ++e) { m_r[e] = -1e30f; l_r[e] = 0.f; }
#pragma unroll
    for (int j = 0; j < 4; ++j) o[j] = (f32x8){0, 0, 0, 0, 0, 0, 0, 0};

    const float scale = 0.125f;  // 1/sqrt(64)
    const int nsteps = (q0 + 16 + 31) / 32;

    for (int st = 0; st < nsteps; ++st) {
        const int k0 = st * 32;

        // Scores for two 16-key sub-tiles.
        f32x8 s0 = (f32x8){0, 0, 0, 0, 0, 0, 0, 0};
        f32x8 s1 = (f32x8){0, 0, 0, 0, 0, 0, 0, 0};
        {
            Frag ka, kb2;
            const u16* Krow = Kp + (size_t)(k0 + r) * D_;
            ka.q[0] = *(const u32x4*)(Krow + half * 8);
            ka.q[1] = *(const u32x4*)(Krow + 16 + half * 8);
            kb2.q[0] = *(const u32x4*)(Krow + 32 + half * 8);
            kb2.q[1] = *(const u32x4*)(Krow + 48 + half * 8);
            s0 = wmma_bf16(qf[0].v, ka.v, s0);
            s0 = wmma_bf16(qf[1].v, kb2.v, s0);
            Krow = Kp + (size_t)(k0 + 16 + r) * D_;
            ka.q[0] = *(const u32x4*)(Krow + half * 8);
            ka.q[1] = *(const u32x4*)(Krow + 16 + half * 8);
            kb2.q[0] = *(const u32x4*)(Krow + 32 + half * 8);
            kb2.q[1] = *(const u32x4*)(Krow + 48 + half * 8);
            s1 = wmma_bf16(qf[0].v, ka.v, s1);
            s1 = wmma_bf16(qf[1].v, kb2.v, s1);
        }

        // Online softmax; rows live across 16 lanes of one half-wave.
        float corr[8];
#pragma unroll
        for (int e = 0; e < 8; ++e) {
            const int qg = q0 + e + half * 8;
            float a = s0[e] * scale;
            float bb = s1[e] * scale;
            if (k0 + r > qg) a = -1e30f;
            if (k0 + 16 + r > qg) bb = -1e30f;
            float mx = fmaxf(a, bb);
#pragma unroll
            for (int mm = 8; mm >= 1; mm >>= 1)
                mx = fmaxf(mx, __shfl_xor(mx, mm, 32));
            const float nm = fmaxf(m_r[e], mx);
            const float c = expf(m_r[e] - nm);
            const float p0 = expf(a - nm);
            const float p1 = expf(bb - nm);
            float rs2 = p0 + p1;
#pragma unroll
            for (int mm = 8; mm >= 1; mm >>= 1) rs2 += __shfl_xor(rs2, mm, 32);
            l_r[e] = l_r[e] * c + rs2;
            m_r[e] = nm;
            corr[e] = c;
            Pl[w][e + half * 8][r] = f32_to_bf16(p0);
            Pl[w][e + half * 8][16 + r] = f32_to_bf16(p1);
        }

        // Rescale running output.
#pragma unroll
        for (int j = 0; j < 4; ++j)
#pragma unroll
            for (int e = 0; e < 8; ++e) o[j][e] *= corr[e];

        // P as A-fragment (same-wave LDS RAW stays in order).
        Frag pf;
        pf.q[0] = *(const u32x4*)&Pl[w][r][half * 8];
        pf.q[1] = *(const u32x4*)&Pl[w][r][16 + half * 8];

        // V as B-fragments straight from V^T: lane = dh col, elements = keys.
#pragma unroll
        for (int j = 0; j < 4; ++j) {
            Frag vf;
            const u16* vrow = Vth + (size_t)(j * 16 + r) * S_ + k0;
            vf.q[0] = *(const u32x4*)(vrow + half * 8);
            vf.q[1] = *(const u32x4*)(vrow + 16 + half * 8);
            o[j] = wmma_bf16(pf.v, vf.v, o[j]);
        }
    }

    // Normalize and store ctx (bf16), [B,S,D] with head offset.
#pragma unroll
    for (int j = 0; j < 4; ++j) {
        const int col = h * DH_ + j * 16 + r;
#pragma unroll
        for (int e = 0; e < 8; ++e) {
            const int row = q0 + e + half * 8;
            float v = o[j][e] / l_r[e];
            Ctx[((size_t)b * S_ + row) * D_ + col] = f32_to_bf16(v);
        }
    }
}

// ---------------------------------------------------------------------------
// Launch
// ---------------------------------------------------------------------------
extern "C" void kernel_launch(void* const* d_in, const int* in_sizes, int n_in,
                              void* d_out, int out_size, void* d_ws,
                              size_t ws_size, hipStream_t stream) {
    (void)in_sizes; (void)n_in; (void)out_size; (void)ws_size;

    const float* x    = (const float*)d_in[0];
    const float* Wq   = (const float*)d_in[1];
    const float* Wk   = (const float*)d_in[2];
    const float* Wv   = (const float*)d_in[3];
    const float* Wo   = (const float*)d_in[4];
    const float* bo   = (const float*)d_in[5];
    const float* W1   = (const float*)d_in[6];
    const float* b1   = (const float*)d_in[7];
    const float* W2   = (const float*)d_in[8];
    const float* b2   = (const float*)d_in[9];
    const float* ln1s = (const float*)d_in[10];
    const float* ln1b = (const float*)d_in[11];
    const float* ln2s = (const float*)d_in[12];
    const float* ln2b = (const float*)d_in[13];
    float* out = (float*)d_out;

    char* ws = (char*)d_ws;
    size_t off = 0;
    auto carve = [&](size_t bytes) -> char* {
        char* p = ws + off;
        off += (bytes + 255) & ~(size_t)255;
        return p;
    };
    u16* wq_bf  = (u16*)carve((size_t)D_ * D_ * 2);
    u16* wk_bf  = (u16*)carve((size_t)D_ * D_ * 2);
    u16* wv_bf  = (u16*)carve((size_t)D_ * D_ * 2);
    u16* wo_bf  = (u16*)carve((size_t)D_ * D_ * 2);
    u16* w1_bf  = (u16*)carve((size_t)4 * D_ * D_ * 2);
    u16* w2_bf  = (u16*)carve((size_t)4 * D_ * D_ * 2);
    u16* xn_bf  = (u16*)carve((size_t)BS_ * D_ * 2);
    u16* q_bf   = (u16*)carve((size_t)BS_ * D_ * 2);
    u16* k_bf   = (u16*)carve((size_t)BS_ * D_ * 2);
    u16* vt_bf  = (u16*)carve((size_t)BS_ * D_ * 2);  // V^T [B,H,DH,S]
    u16* ctx_bf = (u16*)carve((size_t)BS_ * D_ * 2);
    float* h_f  = (float*)carve((size_t)BS_ * D_ * 4);
    u16* y2_bf  = (u16*)carve((size_t)BS_ * D_ * 2);
    u16* t_bf   = (u16*)carve((size_t)BS_ * 4 * D_ * 2);

    // 1) weights -> bf16
    cvt_f32_bf16_kernel<<<1024, 256, 0, stream>>>(Wq, wq_bf, D_ * D_);
    cvt_f32_bf16_kernel<<<1024, 256, 0, stream>>>(Wk, wk_bf, D_ * D_);
    cvt_f32_bf16_kernel<<<1024, 256, 0, stream>>>(Wv, wv_bf, D_ * D_);
    cvt_f32_bf16_kernel<<<1024, 256, 0, stream>>>(Wo, wo_bf, D_ * D_);
    cvt_f32_bf16_kernel<<<2048, 256, 0, stream>>>(W1, w1_bf, 4 * D_ * D_);
    cvt_f32_bf16_kernel<<<2048, 256, 0, stream>>>(W2, w2_bf, 4 * D_ * D_);

    // 2) ln1(x) -> bf16
    layernorm_bf16_kernel<<<BS_, 256, 0, stream>>>(x, ln1s, ln1b, xn_bf);

    // 3) Q/K/V projections
    dim3 g1(D_ / 128, BS_ / 128);
    gemm_bf16_kernel<false, false, false, false, true, false>
        <<<g1, 256, 0, stream>>>(xn_bf, wq_bf, nullptr, nullptr, nullptr, q_bf,
                                 BS_, D_, D_);
    gemm_bf16_kernel<false, false, false, false, true, false>
        <<<g1, 256, 0, stream>>>(xn_bf, wk_bf, nullptr, nullptr, nullptr, k_bf,
                                 BS_, D_, D_);
    gemm_bf16_kernel<false, false, false, false, false, true>
        <<<g1, 256, 0, stream>>>(xn_bf, wv_bf, nullptr, nullptr, nullptr, vt_bf,
                                 BS_, D_, D_);

    // 4) causal flash attention -> ctx (bf16)
    flash_attn_kernel<<<dim3(S_ / 64, H_, B_), 128, 0, stream>>>(q_bf, k_bf,
                                                                 vt_bf, ctx_bf);

    // 5) out-proj + bias + residual(x) -> h (f32)
    gemm_bf16_kernel<true, false, true, true, false, false>
        <<<g1, 256, 0, stream>>>(ctx_bf, wo_bf, bo, x, h_f, nullptr,
                                 BS_, D_, D_);

    // 6) ln2(h) -> bf16
    layernorm_bf16_kernel<<<BS_, 256, 0, stream>>>(h_f, ln2s, ln2b, y2_bf);

    // 7) MLP up + bias + GELU -> t (bf16)
    dim3 g2(4 * D_ / 128, BS_ / 128);
    gemm_bf16_kernel<true, true, false, false, true, false>
        <<<g2, 256, 0, stream>>>(y2_bf, w1_bf, b1, nullptr, nullptr, t_bf,
                                 BS_, 4 * D_, D_);

    // 8) MLP down + bias + residual(h) -> out (f32)
    gemm_bf16_kernel<true, false, true, true, false, false>
        <<<g1, 256, 0, stream>>>(t_bf, w2_bf, b2, h_f, out, nullptr,
                                 BS_, D_, 4 * D_);
}
